// CTM_790273982469
// MI455X (gfx1250) — matrix operand order repaired and checked by reading, not compiled, not run on hardware
//
#include <hip/hip_runtime.h>

// ---------------------------------------------------------------------------
// CTM log-prob: out = s + mu + RHO * s @ theta_off^T,  s = x @ beta^T
//   x: [2048, 50000] f32, beta: [64, 50000] f32, theta: [64,64], mu: [64]
//
// Split-precision bf16 WMMA (x = xh+xl, b = bh+bl; keep xh*bh + xh*bl + xl*bh):
// fp32-class accuracy on the bf16 matrix pipe, the only unit that clears the
// ~723 TFLOP/s needed to hold the 23.3 TB/s HBM roofline (~18us) at this
// problem's 31 FLOP/byte. beta's hi/lo split is precomputed once (12.8 MB,
// L2-resident) so the hot loop converts only the 16 x-floats per chunk, with
// v_perm_b32 doing the bf16 pair-packing in a single VALU op.
// ---------------------------------------------------------------------------

#define B_ROWS    2048
#define VOCAB     50000
#define NTOP      64
#define RHO_C     0.1f
#define KSPLIT    4
#define SPLIT_LEN 12512          // 391*32; last split = 12464 (= 389*32 + 16)
#define WAVES_PB  8              // 8 wave32 per block

typedef __bf16        v16bf __attribute__((ext_vector_type(16)));
typedef float         v8f   __attribute__((ext_vector_type(8)));
typedef float         v4f   __attribute__((ext_vector_type(4)));
typedef unsigned int  v8u   __attribute__((ext_vector_type(8)));
typedef unsigned int  v4u   __attribute__((ext_vector_type(4)));
typedef unsigned int  v2u   __attribute__((ext_vector_type(2)));

// Pack two floats' bf16-truncations into one dword with a single v_perm_b32:
// result = { u1[31:16], u0[31:16] }.
static __device__ __forceinline__ unsigned pack_hi(unsigned u0, unsigned u1) {
    return __builtin_amdgcn_perm(u1, u0, 0x07060302u);
}

// Truncation split of 16 floats into packed hi/lo bf16 fragments.
// f = hi + lo + O(2^-17 |f|); perm/sub ops co-execute with the XDL WMMA pipe.
static __device__ __forceinline__ void split16(const float* f, v16bf& h, v16bf& l) {
    v8u hu, lu;
#pragma unroll
    for (int i = 0; i < 8; ++i) {
        const float f0 = f[2 * i], f1 = f[2 * i + 1];
        const unsigned u0 = __float_as_uint(f0), u1 = __float_as_uint(f1);
        hu[i] = pack_hi(u0, u1);
        const float l0 = f0 - __uint_as_float(u0 & 0xFFFF0000u);
        const float l1 = f1 - __uint_as_float(u1 & 0xFFFF0000u);
        lu[i] = pack_hi(__float_as_uint(l0), __float_as_uint(l1));
    }
    h = __builtin_bit_cast(v16bf, hu);
    l = __builtin_bit_cast(v16bf, lu);
}

// 8 consecutive floats (two aligned float4, nontemporal: x is single-pass).
template <bool GUARD>
static __device__ __forceinline__ void loadA8(float* d, const float* __restrict__ row,
                                              int k, int kend) {
    if (!GUARD || k < kend) {
        v4f a = __builtin_nontemporal_load((const v4f*)(row + k));
        v4f b = __builtin_nontemporal_load((const v4f*)(row + k + 4));
#pragma unroll
        for (int i = 0; i < 4; ++i) { d[i] = a[i]; d[4 + i] = b[i]; }
    } else {
#pragma unroll
        for (int i = 0; i < 8; ++i) d[i] = 0.0f;
    }
}

// 16 consecutive packed bf16 (two aligned 16B loads). kend % 16 == 0.
template <bool GUARD>
static __device__ __forceinline__ v16bf loadB16(const __bf16* __restrict__ row,
                                                int k, int kend) {
    union { v16bf bf; v4u u[2]; } b;
    if (!GUARD || k < kend) {
        b.u[0] = *(const v4u*)(row + k);
        b.u[1] = *(const v4u*)(row + k + 8);
    } else {
        b.u[0] = v4u{0, 0, 0, 0};
        b.u[1] = v4u{0, 0, 0, 0};
    }
    return b.bf;
}

static __device__ __forceinline__ v8f wmma_bf16(v16bf a, v16bf b, v8f c) {
    return __builtin_amdgcn_wmma_f32_16x16x32_bf16(false, a, false, b, (short)0, c,
                                                   false, false);
}

// One 32-wide K chunk: A-fragment split + 4 N-tiles x 3 WMMAs.
template <bool GUARD>
static __device__ __forceinline__ void do_chunk(
    const float* __restrict__ xrow, const __bf16* const* bhr,
    const __bf16* const* blr, int kb, int kend, int hl, v8f* acc) {
    // A (x tile 16x32): lane half -> contiguous K h*16..+15
    float fa[16];
    loadA8<GUARD>(fa + 0, xrow, kb + hl * 16, kend);
    loadA8<GUARD>(fa + 8, xrow, kb + hl * 16 + 8, kend);
    v16bf ah, al;
    split16(fa, ah, al);

    const int kB = kb + hl * 16;   // B: lane half -> contiguous K h*16..+15
#pragma unroll
    for (int t = 0; t < 4; ++t) {
        v16bf bh = loadB16<GUARD>(bhr[t], kB, kend);
        v16bf bl = loadB16<GUARD>(blr[t], kB, kend);
        acc[t] = wmma_bf16(ah, bh, acc[t]);
        acc[t] = wmma_bf16(ah, bl, acc[t]);
        acc[t] = wmma_bf16(al, bh, acc[t]);
    }
}

// ---------------------------------------------------------------------------
// Kernel 0: one-shot beta hi/lo split (64x50000 f32 -> 2x bf16 arrays).
// 800000 threads x 4 elements; exact cover, no bounds check needed.
// ---------------------------------------------------------------------------
__global__ __launch_bounds__(256) void beta_split_kernel(
    const float* __restrict__ beta, unsigned int* __restrict__ bh,
    unsigned int* __restrict__ bl) {
    const size_t i = ((size_t)blockIdx.x * 256 + threadIdx.x) * 4;
    v4f f = *(const v4f*)(beta + i);
    unsigned u[4], lo[4];
#pragma unroll
    for (int j = 0; j < 4; ++j) {
        u[j]  = __float_as_uint(f[j]);
        lo[j] = __float_as_uint(f[j] - __uint_as_float(u[j] & 0xFFFF0000u));
    }
    *(v2u*)(bh + i / 2) = v2u{pack_hi(u[0], u[1]), pack_hi(u[2], u[3])};
    *(v2u*)(bl + i / 2) = v2u{pack_hi(lo[0], lo[1]), pack_hi(lo[2], lo[3])};
}

// ---------------------------------------------------------------------------
// Kernel 1: partial s tiles. grid = (128 M-tiles, 4 K-splits), 256 threads.
// Each wave owns a 16x64 tile; 8 waves stride K in 32-wide chunks.
// ---------------------------------------------------------------------------
__global__ __launch_bounds__(256) void ctm_gemm_kernel(
    const float* __restrict__ x, const __bf16* __restrict__ bhi,
    const __bf16* __restrict__ blo, float* __restrict__ s_part) {

    const int m0    = blockIdx.x * 16;
    const int split = blockIdx.y;
    const int kbeg  = split * SPLIT_LEN;
    const int kend  = min(VOCAB, kbeg + SPLIT_LEN);

    const int lane = threadIdx.x & 31;
    const int wave = threadIdx.x >> 5;
    const int r    = lane & 15;    // row-in-tile (A) / col-in-tile (B)
    const int hl   = lane >> 4;    // lane half selects K sub-block

    const float* __restrict__ xrow = x + (size_t)(m0 + r) * VOCAB;
    const __bf16* bhr[4];
    const __bf16* blr[4];
#pragma unroll
    for (int t = 0; t < 4; ++t) {
        bhr[t] = bhi + (size_t)(t * 16 + r) * VOCAB;
        blr[t] = blo + (size_t)(t * 16 + r) * VOCAB;
    }

    v8f acc[4] = {v8f{0}, v8f{0}, v8f{0}, v8f{0}};

    int kb = kbeg + wave * 32;
    for (; kb + 32 <= kend; kb += WAVES_PB * 32)      // guard-free steady state
        do_chunk<false>(xrow, bhr, blr, kb, kend, hl, acc);
    if (kb < kend)                                    // single 16-wide tail
        do_chunk<true>(xrow, bhr, blr, kb, kend, hl, acc);

    // ---- deterministic cross-wave reduction through LDS (32 KB) ----
    __shared__ float red[WAVES_PB][16][NTOP];
#pragma unroll
    for (int t = 0; t < 4; ++t)
#pragma unroll
        for (int i = 0; i < 8; ++i)
            red[wave][i + 8 * hl][t * 16 + r] = acc[t][i];
    __syncthreads();

    for (int idx = threadIdx.x; idx < 16 * NTOP; idx += 256) {
        const int row = idx >> 6;
        const int col = idx & (NTOP - 1);
        float v = 0.0f;
#pragma unroll
        for (int w = 0; w < WAVES_PB; ++w) v += red[w][row][col];
        s_part[((size_t)split * B_ROWS + (m0 + row)) * NTOP + col] = v;
    }
}

// ---------------------------------------------------------------------------
// Kernel 2: sum the 4 K-split partials, add mu + RHO * s @ theta_off^T.
// ---------------------------------------------------------------------------
__global__ __launch_bounds__(64) void ctm_finalize_kernel(
    const float* __restrict__ s_part, const float* __restrict__ theta,
    const float* __restrict__ mu, float* __restrict__ out) {

    const int b = blockIdx.x;
    const int k = threadIdx.x;

    __shared__ float srow[NTOP];
    float v = 0.0f;
#pragma unroll
    for (int c = 0; c < KSPLIT; ++c)
        v += s_part[((size_t)c * B_ROWS + b) * NTOP + k];
    srow[k] = v;
    __syncthreads();

    float dot = 0.0f;
#pragma unroll
    for (int j = 0; j < NTOP; ++j) {
        float w = (j == k) ? 0.0f : theta[k * NTOP + j];   // zero diagonal
        dot = fmaf(srow[j], w, dot);
    }
    out[(size_t)b * NTOP + k] = v + mu[k] + RHO_C * dot;
}

// ---------------------------------------------------------------------------
extern "C" void kernel_launch(void* const* d_in, const int* in_sizes, int n_in,
                              void* d_out, int out_size, void* d_ws, size_t ws_size,
                              hipStream_t stream) {
    const float* x     = (const float*)d_in[0];   // [2048, 50000]
    const float* beta  = (const float*)d_in[1];   // [64, 50000]
    const float* theta = (const float*)d_in[2];   // [64, 64]
    const float* mu    = (const float*)d_in[3];   // [64]
    float* out = (float*)d_out;                   // [2048, 64]

    // workspace layout: spart (2 MB) | beta_hi (6.4 MB) | beta_lo (6.4 MB)
    float*  spart = (float*)d_ws;
    __bf16* bhi   = (__bf16*)(spart + (size_t)KSPLIT * B_ROWS * NTOP);
    __bf16* blo   = bhi + (size_t)NTOP * VOCAB;

    beta_split_kernel<<<(NTOP * VOCAB) / (256 * 4), 256, 0, stream>>>(
        beta, (unsigned int*)bhi, (unsigned int*)blo);
    ctm_gemm_kernel<<<dim3(B_ROWS / 16, KSPLIT), 256, 0, stream>>>(x, bhi, blo, spart);
    ctm_finalize_kernel<<<B_ROWS, NTOP, 0, stream>>>(spart, theta, mu, out);
}
